// Model_84911503442313
// MI455X (gfx1250) — matrix operand order repaired and checked by reading, not compiled, not run on hardware
//
#include <hip/hip_runtime.h>
#include <hip/hip_bf16.h>

// ---------------------------------------------------------------------------
// Model: line-classifier over attention features.
// B=4, S=2048, H=768, NH=8, DH=96, L=128, ML=15.  All GEMM-shaped math runs
// through v_wmma_f32_16x16x32_f16 (wave32, fp32 accumulate).
// GEMMs are register-blocked 16x64 per wave (A-fragment reused 4x per K-step)
// with 4 waves per block (64x64 block tile).
// ---------------------------------------------------------------------------

#define Bq   4
#define Sq   2048
#define Hq   768
#define NHq  8
#define DHq  96
#define Lq   128
#define MLq  15

typedef __attribute__((ext_vector_type(16))) _Float16 v16h;
typedef __attribute__((ext_vector_type(8)))  _Float16 v8h;
typedef __attribute__((ext_vector_type(8)))  float    v8f;

__device__ __forceinline__ v16h joinh(v8h a, v8h b) {
    v16h r;
#pragma unroll
    for (int i = 0; i < 8; ++i) { r[i] = a[i]; r[i + 8] = b[i]; }
    return r;
}

// ---------------------------------------------------------------------------
// fp32 -> f16 conversion
// ---------------------------------------------------------------------------
__global__ __launch_bounds__(256) void cvt_f16_kernel(const float* __restrict__ in,
                                                      _Float16* __restrict__ out, int n) {
    int i = blockIdx.x * 256 + threadIdx.x;
    if (i < n) out[i] = (_Float16)in[i];
}

// ---------------------------------------------------------------------------
// WMMA GEMM:  Y[M,N] = X[M,K] @ W[N,K]^T + bias
// Block = 128 threads = 4 waves stacked along M; each wave computes a
// 16x64 strip (4 accumulators, A fragment reused 4x per 32-wide K step).
// blockIdx.x = 64-wide N tile, blockIdx.y = 64-high M tile.
//
// IN_MODE  0: X is f16 row-major [M,K]
//          1: rows gathered from hidden_h via line_starts (Q projection);
//             row = ((b*L + l)*16 + m), token pos = clip(start + m, 0, S-1)
// OUT_MODE 0: f32 row-major [M,N]
//          2: f16 scatter to K layout [B,NH,S,DH]   (row=(b,s), col=(h,d))
//          3: f16 scatter to Vt layout [B,NH,DH,S]
//          4: f16 row-major [M,N] with pad row (m==15) forced to zero (Q)
// ---------------------------------------------------------------------------
template <int IN_MODE, int OUT_MODE>
__global__ __launch_bounds__(128)
void gemm_wmma_kernel(const _Float16* __restrict__ X,
                      const _Float16* __restrict__ W,
                      const float* __restrict__ bias,
                      float* __restrict__ outf,
                      _Float16* __restrict__ outh,
                      int M, int N, int K,
                      const int* __restrict__ line_starts,
                      const _Float16* __restrict__ hidden_h) {
    const int tid  = threadIdx.x;
    const int wave = tid >> 5;                 // 0..3 : M sub-tile within block
    const int lane = tid & 31;
    const int nsub = lane & 15;
    const int half = lane >> 4;

    const int tn64 = blockIdx.x;               // 64-wide N strip
    const int tm   = blockIdx.y * 4 + wave;    // 16-high M tile

    // A-operand row for this lane (both halves cover the same 16 rows,
    // halves select different K sub-ranges per the 16-bit A layout).
    const int arow = tm * 16 + nsub;
    const _Float16* xrow;
    if (IN_MODE == 0) {
        xrow = X + (size_t)arow * K;
    } else {
        const int m16 = arow & 15;
        const int bl  = arow >> 4;
        const int b   = bl / Lq;
        const int l   = bl - b * Lq;
        int pos = line_starts[b * Lq + l] + m16;
        pos = pos < 0 ? 0 : (pos > Sq - 1 ? Sq - 1 : pos);
        xrow = hidden_h + (size_t)(b * Sq + pos) * K;
    }

    // Four B-operand column pointers (lane's column within each 16-wide tile).
    const _Float16* wrow[4];
#pragma unroll
    for (int j = 0; j < 4; ++j)
        wrow[j] = W + (size_t)(tn64 * 64 + j * 16 + nsub) * K;

    v8f acc[4] = {};
    for (int kc = 0; kc < K; kc += 32) {
        // A fragment, loaded once and reused for all 4 N tiles.
        v8h a0 = *(const v8h*)(xrow + kc + half * 16);
        v8h a1 = *(const v8h*)(xrow + kc + half * 16 + 8);
        v16h afrag = joinh(a0, a1);
        __builtin_prefetch(xrow + kc + 128, 0, 0);   // global_prefetch_b8
#pragma unroll
        for (int j = 0; j < 4; ++j) {
            v16h bfrag = *(const v16h*)(wrow[j] + kc + half * 16);
            acc[j] = __builtin_amdgcn_wmma_f32_16x16x32_f16(false, afrag, false, bfrag,
                                                            (short)0, acc[j], false, false);
        }
        __builtin_prefetch(wrow[0] + kc + 128, 0, 0);
    }

#pragma unroll
    for (int j = 0; j < 4; ++j) {
        const int col = tn64 * 64 + j * 16 + nsub;
        const float bv = bias ? bias[col] : 0.0f;
#pragma unroll
        for (int r = 0; r < 8; ++r) {
            const int row = tm * 16 + r + 8 * half;   // C layout: M = r + 8*half
            float v = acc[j][r] + bv;
            if (OUT_MODE == 0) {
                outf[(size_t)row * N + col] = v;
            } else if (OUT_MODE == 2) {               // K: [B,NH,S,DH]
                const int b = row >> 11, s = row & (Sq - 1);
                const int h = col / DHq, d = col - h * DHq;
                outh[(((size_t)(b * NHq + h) * Sq) + s) * DHq + d] = (_Float16)v;
            } else if (OUT_MODE == 3) {               // Vt: [B,NH,DH,S]
                const int b = row >> 11, s = row & (Sq - 1);
                const int h = col / DHq, d = col - h * DHq;
                outh[(((size_t)(b * NHq + h) * DHq) + d) * Sq + s] = (_Float16)v;
            } else {                                  // Q padded, zero pad row
                const int m16 = row & 15;
                outh[(size_t)row * N + col] = (m16 == MLq) ? (_Float16)0.0f : (_Float16)v;
            }
        }
    }
}

// ---------------------------------------------------------------------------
// Flash attention, 1 wave per (b, l, h).
// Q tile: 16 (padded) queries x 96.  Keys processed in chunks of 32.
// scores = Q @ K^T (6 WMMAs / chunk), online softmax in C-fragment layout,
// P relayout via LDS bounce, O += P @ V (6 WMMAs / chunk).
// ---------------------------------------------------------------------------
__global__ __launch_bounds__(32)
void attn_kernel(const _Float16* __restrict__ qbuf,   // [B,L,16,H] f16
                 const _Float16* __restrict__ kbuf,   // [B,NH,S,DH] f16
                 const _Float16* __restrict__ vtbuf,  // [B,NH,DH,S] f16
                 _Float16* __restrict__ obuf,         // [B,L,16,H] f16
                 const int* __restrict__ context_len) {
    __shared__ __align__(32) _Float16 pls[16 * 32];

    const int lane = threadIdx.x;
    const int nsub = lane & 15;
    const int half = lane >> 4;

    const int idx = blockIdx.x;                // b*L*NH + l*NH + h
    const int h  = idx & (NHq - 1);
    const int bl = idx >> 3;
    const int b  = bl / Lq;

    const int clen = context_len[b];
    const float scale = 0.10206207261596575f;  // 1/sqrt(96)

    const _Float16* qb  = qbuf  + ((size_t)bl * 16) * Hq + h * DHq;
    const _Float16* kb  = kbuf  + ((size_t)(b * NHq + h) * Sq) * DHq;
    const _Float16* vtb = vtbuf + ((size_t)(b * NHq + h) * DHq) * Sq;

    // Q as three A-fragments (16x32 each covering feature chunks).
    v16h qa[3];
#pragma unroll
    for (int c = 0; c < 3; ++c) {
        const _Float16* p = qb + (size_t)nsub * Hq + c * 32 + half * 16;
        qa[c] = joinh(*(const v8h*)p, *(const v8h*)(p + 8));
    }

    float rmax[8], rsum[8];
    v8f oacc[6];
    const v8f zero = {};
#pragma unroll
    for (int r = 0; r < 8; ++r) { rmax[r] = -1e30f; rsum[r] = 0.0f; }
#pragma unroll
    for (int j = 0; j < 6; ++j) oacc[j] = zero;

    for (int s0 = 0; s0 < Sq; s0 += 32) {
        // --- scores for 32 keys: two 16-key C fragments ---
        v8f sc0 = zero, sc1 = zero;
#pragma unroll
        for (int c = 0; c < 3; ++c) {
            v16h bk0 = *(const v16h*)(kb + (size_t)(s0 + nsub) * DHq + c * 32 + half * 16);
            v16h bk1 = *(const v16h*)(kb + (size_t)(s0 + 16 + nsub) * DHq + c * 32 + half * 16);
            sc0 = __builtin_amdgcn_wmma_f32_16x16x32_f16(false, qa[c], false, bk0,
                                                         (short)0, sc0, false, false);
            sc1 = __builtin_amdgcn_wmma_f32_16x16x32_f16(false, qa[c], false, bk1,
                                                         (short)0, sc1, false, false);
        }

        const int key0 = s0 + nsub;
        const int key1 = s0 + 16 + nsub;
        const bool ok0 = key0 < clen;
        const bool ok1 = key1 < clen;

        // --- online softmax per row (row = r + 8*half, shared by 16 lanes) ---
#pragma unroll
        for (int r = 0; r < 8; ++r) {
            float a0 = ok0 ? sc0[r] * scale : -1e9f;
            float a1 = ok1 ? sc1[r] * scale : -1e9f;
            float cm = fmaxf(a0, a1);
#pragma unroll
            for (int mk = 1; mk < 16; mk <<= 1) cm = fmaxf(cm, __shfl_xor(cm, mk, 32));
            const float nm   = fmaxf(rmax[r], cm);
            const float corr = __expf(rmax[r] - nm);
            const float p0   = __expf(a0 - nm);
            const float p1   = __expf(a1 - nm);
            float ps = p0 + p1;
#pragma unroll
            for (int mk = 1; mk < 16; mk <<= 1) ps += __shfl_xor(ps, mk, 32);
            rsum[r] = rsum[r] * corr + ps;
            rmax[r] = nm;
#pragma unroll
            for (int j = 0; j < 6; ++j) oacc[j][r] *= corr;

            const int row = r + 8 * half;
            pls[row * 32 + nsub]      = (_Float16)p0;
            pls[row * 32 + 16 + nsub] = (_Float16)p1;
        }
        __syncthreads();

        // --- P fragment in A layout (16 rows x 32 keys) ---
        v8h pa0 = *(const v8h*)(pls + nsub * 32 + half * 16);
        v8h pa1 = *(const v8h*)(pls + nsub * 32 + half * 16 + 8);
        v16h pa = joinh(pa0, pa1);
        __syncthreads();

        // --- O += P @ V, six 16-wide DH chunks ---
#pragma unroll
        for (int j = 0; j < 6; ++j) {
            v16h vb = *(const v16h*)(vtb + (size_t)(j * 16 + nsub) * Sq + s0 + half * 16);
            oacc[j] = __builtin_amdgcn_wmma_f32_16x16x32_f16(false, pa, false, vb,
                                                             (short)0, oacc[j], false, false);
        }
    }

    _Float16* ob = obuf + ((size_t)bl * 16) * Hq + h * DHq;
#pragma unroll
    for (int j = 0; j < 6; ++j) {
#pragma unroll
        for (int r = 0; r < 8; ++r) {
            const int row = r + 8 * half;
            const float den = rsum[r] > 0.0f ? rsum[r] : 1.0f;
            ob[(size_t)row * Hq + j * 16 + nsub] = (_Float16)(oacc[j][r] / den);
        }
    }
}

// ---------------------------------------------------------------------------
// Masked mean-pool:  x = concat(line_mean, attn_mean)  [B*L, 2H]  (f32 + f16)
// ---------------------------------------------------------------------------
__global__ __launch_bounds__(256)
void pool_kernel(const float* __restrict__ hidden,     // [B,S,H] f32
                 const float* __restrict__ attn_o,     // [B*L*16, H] f32
                 const int* __restrict__ line_starts,
                 const int* __restrict__ line_lens,
                 float* __restrict__ xcat,             // [B*L, 2H] f32
                 _Float16* __restrict__ xcath) {       // [B*L, 2H] f16
    const int bl = blockIdx.x;
    const int b = bl / Lq;
    const int len = line_lens[bl];
    const int start = line_starts[bl];
    const float inv = 1.0f / (float)(len > 0 ? len : 1);
    for (int c = threadIdx.x; c < Hq; c += 256) {
        float sl = 0.0f, sa = 0.0f;
        for (int m = 0; m < MLq; ++m) {
            if (m < len) {
                int pos = start + m;
                pos = pos < 0 ? 0 : (pos > Sq - 1 ? Sq - 1 : pos);
                sl += hidden[(size_t)(b * Sq + pos) * Hq + c];
                sa += attn_o[((size_t)bl * 16 + m) * Hq + c];
            }
        }
        const float vl = sl * inv, va = sa * inv;
        xcat[(size_t)bl * (2 * Hq) + c]       = vl;
        xcat[(size_t)bl * (2 * Hq) + Hq + c]  = va;
        xcath[(size_t)bl * (2 * Hq) + c]      = (_Float16)vl;
        xcath[(size_t)bl * (2 * Hq) + Hq + c] = (_Float16)va;
    }
}

// ---------------------------------------------------------------------------
// LayerNorm -> GELU (exact) [+ residual].  One 256-thread block per row.
// ---------------------------------------------------------------------------
__global__ __launch_bounds__(256)
void ln_gelu_kernel(const float* __restrict__ y, const float* __restrict__ g,
                    const float* __restrict__ be, const float* __restrict__ resid,
                    float* __restrict__ outf, _Float16* __restrict__ outh) {
    const int row = blockIdx.x;
    const int tid = threadIdx.x;
    __shared__ float r1[256], r2[256];
    const float* yr = y + (size_t)row * Hq;
    float s = 0.0f, s2 = 0.0f;
    for (int c = tid; c < Hq; c += 256) { float v = yr[c]; s += v; s2 += v * v; }
    r1[tid] = s; r2[tid] = s2;
    __syncthreads();
    for (int st = 128; st > 0; st >>= 1) {
        if (tid < st) { r1[tid] += r1[tid + st]; r2[tid] += r2[tid + st]; }
        __syncthreads();
    }
    const float mean = r1[0] * (1.0f / (float)Hq);
    const float var  = r2[0] * (1.0f / (float)Hq) - mean * mean;
    const float rstd = rsqrtf(var + 1e-5f);
    for (int c = tid; c < Hq; c += 256) {
        const float t  = (yr[c] - mean) * rstd * g[c] + be[c];
        const float ge = 0.5f * t * (1.0f + erff(t * 0.70710678118654752f));
        const float o  = resid ? (resid[(size_t)row * Hq + c] + ge) : ge;
        outf[(size_t)row * Hq + c] = o;
        if (outh) outh[(size_t)row * Hq + c] = (_Float16)o;
    }
}

// ---------------------------------------------------------------------------
// Final head: logit = x2 . W3 + b3 ; masked -10 for empty lines ; sigmoid.
// ---------------------------------------------------------------------------
__global__ __launch_bounds__(256)
void head_kernel(const float* __restrict__ x2, const float* __restrict__ W3,
                 const float* __restrict__ b3, const int* __restrict__ line_lens,
                 float* __restrict__ out) {
    const int row = blockIdx.x;
    __shared__ float red[256];
    float s = 0.0f;
    for (int c = threadIdx.x; c < Hq; c += 256) s += x2[(size_t)row * Hq + c] * W3[c];
    red[threadIdx.x] = s;
    __syncthreads();
    for (int st = 128; st > 0; st >>= 1) {
        if (threadIdx.x < st) red[threadIdx.x] += red[threadIdx.x + st];
        __syncthreads();
    }
    if (threadIdx.x == 0) {
        float logit = red[0] + b3[0];
        if (line_lens[row] <= 0) logit = -10.0f;
        out[row] = 1.0f / (1.0f + __expf(-logit));
    }
}

// ---------------------------------------------------------------------------
// Launch
// ---------------------------------------------------------------------------
extern "C" void kernel_launch(void* const* d_in, const int* in_sizes, int n_in,
                              void* d_out, int out_size, void* d_ws, size_t ws_size,
                              hipStream_t stream) {
    const float* hidden     = (const float*)d_in[0];
    const float* in_proj_w  = (const float*)d_in[1];
    const float* in_proj_b  = (const float*)d_in[2];
    const float* out_proj_w = (const float*)d_in[3];
    const float* out_proj_b = (const float*)d_in[4];
    const float* W1  = (const float*)d_in[5];
    const float* b1  = (const float*)d_in[6];
    const float* g1  = (const float*)d_in[7];
    const float* be1 = (const float*)d_in[8];
    const float* W2  = (const float*)d_in[9];
    const float* b2  = (const float*)d_in[10];
    const float* g2  = (const float*)d_in[11];
    const float* be2 = (const float*)d_in[12];
    const float* W3  = (const float*)d_in[13];
    const float* b3  = (const float*)d_in[14];
    const int* line_starts = (const int*)d_in[15];
    const int* line_lens   = (const int*)d_in[16];
    const int* context_len = (const int*)d_in[17];
    float* out = (float*)d_out;

    char* p = (char*)d_ws;
    auto carve = [&](size_t bytes) -> void* {
        void* r = (void*)p;
        p += (bytes + 255) & ~(size_t)255;
        return r;
    };

    const size_t nHid  = (size_t)Bq * Sq * Hq;        // 6,291,456
    const size_t nRowQ = (size_t)Bq * Lq * 16;        // 8,192 padded line rows
    const size_t nBL   = (size_t)Bq * Lq;             // 512

    _Float16* hidden_h = (_Float16*)carve(nHid * 2);
    _Float16* inproj_h = (_Float16*)carve((size_t)3 * Hq * Hq * 2);
    _Float16* wo_h     = (_Float16*)carve((size_t)Hq * Hq * 2);
    _Float16* w1_h     = (_Float16*)carve((size_t)Hq * 2 * Hq * 2);
    _Float16* w2_h     = (_Float16*)carve((size_t)Hq * Hq * 2);
    _Float16* kbuf     = (_Float16*)carve(nHid * 2);            // [B,NH,S,DH]
    _Float16* vtbuf    = (_Float16*)carve(nHid * 2);            // [B,NH,DH,S]
    _Float16* qbuf     = (_Float16*)carve(nRowQ * Hq * 2);      // [B,L,16,H]
    _Float16* obuf     = (_Float16*)carve(nRowQ * Hq * 2);      // [B,L,16,H]
    float*    attn_o   = (float*)carve(nRowQ * Hq * 4);         // f32 [8192,768]
    float*    xcat     = (float*)carve(nBL * 2 * Hq * 4);
    _Float16* xcath    = (_Float16*)carve(nBL * 2 * Hq * 2);
    float*    y1       = (float*)carve(nBL * Hq * 4);
    float*    z        = (float*)carve(nBL * Hq * 4);
    _Float16* zh       = (_Float16*)carve(nBL * Hq * 2);
    float*    y2       = (float*)carve(nBL * Hq * 4);
    float*    x2       = (float*)carve(nBL * Hq * 4);

    // --- 1. fp32 -> f16 conversions ---
    auto cvt = [&](const float* src, _Float16* dst, size_t n) {
        cvt_f16_kernel<<<(unsigned)((n + 255) / 256), 256, 0, stream>>>(src, dst, (int)n);
    };
    cvt(hidden, hidden_h, nHid);
    cvt(in_proj_w, inproj_h, (size_t)3 * Hq * Hq);
    cvt(out_proj_w, wo_h, (size_t)Hq * Hq);
    cvt(W1, w1_h, (size_t)Hq * 2 * Hq);
    cvt(W2, w2_h, (size_t)Hq * Hq);

    // --- 2. projections (WMMA) ---
    dim3 gBig(Hq / 64, (Bq * Sq) / 64);   // (12, 128)
    gemm_wmma_kernel<0, 2><<<gBig, 128, 0, stream>>>(
        hidden_h, inproj_h + (size_t)Hq * Hq, in_proj_b + Hq,
        nullptr, kbuf, Bq * Sq, Hq, Hq, nullptr, nullptr);
    gemm_wmma_kernel<0, 3><<<gBig, 128, 0, stream>>>(
        hidden_h, inproj_h + (size_t)2 * Hq * Hq, in_proj_b + 2 * Hq,
        nullptr, vtbuf, Bq * Sq, Hq, Hq, nullptr, nullptr);
    dim3 gQ(Hq / 64, (int)(nRowQ / 64));  // (12, 128)
    gemm_wmma_kernel<1, 4><<<gQ, 128, 0, stream>>>(
        nullptr, inproj_h, in_proj_b,
        nullptr, qbuf, (int)nRowQ, Hq, Hq, line_starts, hidden_h);

    // --- 3. flash attention (WMMA) ---
    attn_kernel<<<Bq * Lq * NHq, 32, 0, stream>>>(qbuf, kbuf, vtbuf, obuf, context_len);

    // --- 4. out_proj (WMMA) ---
    gemm_wmma_kernel<0, 0><<<gQ, 128, 0, stream>>>(
        obuf, wo_h, out_proj_b, attn_o, nullptr, (int)nRowQ, Hq, Hq, nullptr, nullptr);

    // --- 5. masked mean pool + concat ---
    pool_kernel<<<(unsigned)nBL, 256, 0, stream>>>(hidden, attn_o, line_starts, line_lens,
                                                   xcat, xcath);

    // --- 6. classifier ---
    dim3 gCls(Hq / 64, (int)(nBL / 64));  // (12, 8)
    gemm_wmma_kernel<0, 0><<<gCls, 128, 0, stream>>>(
        xcath, w1_h, b1, y1, nullptr, (int)nBL, Hq, 2 * Hq, nullptr, nullptr);
    ln_gelu_kernel<<<(unsigned)nBL, 256, 0, stream>>>(y1, g1, be1, nullptr, z, zh);
    gemm_wmma_kernel<0, 0><<<gCls, 128, 0, stream>>>(
        zh, w2_h, b2, y2, nullptr, (int)nBL, Hq, Hq, nullptr, nullptr);
    ln_gelu_kernel<<<(unsigned)nBL, 256, 0, stream>>>(y2, g2, be2, z, x2, nullptr);

    // --- 7. head + sigmoid ---
    head_kernel<<<(unsigned)nBL, 256, 0, stream>>>(x2, W3, b3, line_lens, out);

    (void)in_sizes; (void)n_in; (void)out_size; (void)ws_size;
}
